// AttentionModel_27693949124628
// MI455X (gfx1250) — matrix operand order repaired and checked by reading, not compiled, not run on hardware
//
#include <hip/hip_runtime.h>
#include <hip/hip_bf16.h>

typedef __attribute__((ext_vector_type(16))) _Float16 v16h;
typedef __attribute__((ext_vector_type(8)))  _Float16 v8h;
typedef __attribute__((ext_vector_type(8)))  float    v8f;
typedef _Float16 half_t;

typedef unsigned int u32x4 __attribute__((ext_vector_type(4)));
typedef int          i32x4 __attribute__((ext_vector_type(4)));
typedef int          i32x8 __attribute__((ext_vector_type(8)));

#define BATCH 256
#define NN    200       // nodes
#define NP    208       // nodes padded to 13*16
#define DD    128       // model dim
#define HH    8         // heads
#define KK    16        // key dim per head
#define FFD   512       // feedforward dim
#define KPAD  224       // contraction pad for P@V (multiple of 32 >= 208)
#define ROWS  (BATCH*NN)  // 51200

#if defined(__has_builtin)
#if __has_builtin(__builtin_amdgcn_tensor_load_to_lds) && __has_builtin(__builtin_amdgcn_s_wait_tensorcnt)
#define HAVE_TDM 1
#endif
#endif
#ifndef HAVE_TDM
#define HAVE_TDM 0
#endif

#if HAVE_TDM
// 1-D TDM: copy nelem4 DWORDs from global to LDS via the Tensor Data Mover.
// Descriptor per CDNA5 ISA (type=2, data_size=4B,
// tensor_dim0 = tile_dim0 = nelem4, tensor_dim1 = tile_dim1 = 1).
__device__ __forceinline__ void tdm_load_1d(const void* gptr, void* ldsptr, unsigned nelem4) {
  unsigned long long ga = (unsigned long long)gptr;
  unsigned lds_off = (unsigned)(unsigned long long)ldsptr;  // LDS generic addr low 32 = offset
  u32x4 g0;
  g0[0] = 1u;                                              // count=1 (user, load)
  g0[1] = lds_off;                                         // lds_addr
  g0[2] = (unsigned)ga;                                    // global_addr[31:0]
  g0[3] = (unsigned)((ga >> 32) & 0x01ffffffu) | (2u << 30); // global_addr[56:32] | type=2
  i32x8 g1;
  g1[0] = (int)(2u << 16);                                 // wg_mask=0, data_size=2 (4 bytes)
  g1[1] = (int)((nelem4 & 0xffffu) << 16);                 // tensor_dim0[15:0]
  g1[2] = (int)(((nelem4 >> 16) & 0xffffu) | (1u << 16));  // tensor_dim0[31:16] | tensor_dim1=1
  g1[3] = (int)((nelem4 & 0xffffu) << 16);                 // tile_dim0
  g1[4] = 1;                                               // tile_dim1=1, tile_dim2=0
  g1[5] = (int)nelem4;                                     // tensor_dim0_stride[31:0]
  g1[6] = 0;                                               // stride0 hi | stride1 lo
  g1[7] = 0;
  i32x4 z4 = {0, 0, 0, 0};
#if __clang_major__ >= 23
  i32x8 z8 = {0, 0, 0, 0, 0, 0, 0, 0};
  __builtin_amdgcn_tensor_load_to_lds(g0, g1, z4, z4, z8, 0);
#else
  __builtin_amdgcn_tensor_load_to_lds(g0, g1, z4, z4, 0);
#endif
  __builtin_amdgcn_s_wait_tensorcnt(0);
}
#endif

// ---------------- WMMA fragment helpers (CDNA5 wave32 layouts) ----------------

__device__ __forceinline__ v16h frag_a(const half_t* src, int lda, int m0, int k0, int lane) {
  const half_t* p = src + (size_t)(m0 + (lane & 15)) * lda + k0 + ((lane & 16) ? 16 : 0);
  v8h lo = *(const v8h*)p;
  v8h hi = *(const v8h*)(p + 8);
  return __builtin_shufflevector(lo, hi, 0, 1, 2, 3, 4, 5, 6, 7, 8, 9, 10, 11, 12, 13, 14, 15);
}

__device__ __forceinline__ v16h frag_bt16(const half_t* src, int lds, int k0, int n0, int lane) {
  const half_t* p = src + (size_t)(n0 + (lane & 15)) * lds + k0 + ((lane & 16) ? 16 : 0);
  return *(const v16h*)p;
}

__device__ __forceinline__ v16h frag_bp(const half_t* packed, int ntiles, int kt, int nt, int lane) {
  return ((const v16h*)packed)[((size_t)kt * ntiles + nt) * 32 + lane];
}

__device__ __forceinline__ v8f wmma16(v16h a, v16h b, v8f c) {
  return __builtin_amdgcn_wmma_f32_16x16x32_f16(false, a, false, b, (short)0, c, false, false);
}

__device__ __forceinline__ void store_c_half(half_t* dst, int ldc, int m0, int n0, int lane,
                                             v8f c, float scale) {
  int row0 = m0 + ((lane & 16) ? 8 : 0);
  int col  = n0 + (lane & 15);
#pragma unroll
  for (int r = 0; r < 8; ++r) dst[(size_t)(row0 + r) * ldc + col] = (half_t)(c[r] * scale);
}

__device__ __forceinline__ void store_c_half_T(half_t* dst, int ldc, int m0, int n0, int lane, v8f c) {
  int row0 = m0 + ((lane & 16) ? 8 : 0);
  int col  = n0 + (lane & 15);
#pragma unroll
  for (int r = 0; r < 8; ++r) dst[(size_t)col * ldc + row0 + r] = (half_t)c[r];
}

__device__ __forceinline__ float wave_max(float v) {
#pragma unroll
  for (int o = 16; o > 0; o >>= 1) v = fmaxf(v, __shfl_xor(v, o, 32));
  return v;
}
__device__ __forceinline__ float wave_sum(float v) {
#pragma unroll
  for (int o = 16; o > 0; o >>= 1) v += __shfl_xor(v, o, 32);
  return v;
}

// ---------------- weight pre-swizzle: f32 row-major [K][N] -> fragment-native f16 ----

__global__ void k_pack_b(const float* __restrict__ src, half_t* __restrict__ dst,
                         int K, int N, int nmats) {
  int i = blockIdx.x * 256 + threadIdx.x;
  int per = K * N;
  if (i >= per * nmats) return;
  int mat = i / per;
  int r   = i - mat * per;
  int j    = r & 15;
  int lane = (r >> 4) & 31;
  int t    = r >> 9;
  int ntiles = N >> 4;
  int nt = t % ntiles;
  int kt = t / ntiles;
  int srow = kt * 32 + ((lane & 16) ? 16 : 0) + j;
  int scol = nt * 16 + (lane & 15);
  dst[i] = (half_t)src[(size_t)mat * per + (size_t)srow * N + scol];
}

// ---------------- init embedding ----------------

__global__ void k_init_embed(const float* __restrict__ coords, const float* __restrict__ demand,
                             const float* __restrict__ wd, const float* __restrict__ bd,
                             const float* __restrict__ wi, const float* __restrict__ bi,
                             float* __restrict__ h) {
  int bn = blockIdx.x;
  int n  = bn % NN;
  int d  = threadIdx.x;
  float c0 = coords[(size_t)bn * 2 + 0];
  float c1 = coords[(size_t)bn * 2 + 1];
  float v;
  if (n == 0) v = c0 * wd[d] + c1 * wd[DD + d] + bd[d];
  else        v = c0 * wi[d] + c1 * wi[DD + d] + demand[bn] * wi[2 * DD + d] + bi[d];
  h[(size_t)bn * DD + d] = v;
}

// ---------------- encoder MHA: one block per batch element, all-LDS ----------------

__global__ __launch_bounds__(256) void k_attn(const float* __restrict__ h,
                                              const half_t* __restrict__ wq,
                                              const half_t* __restrict__ wk,
                                              const half_t* __restrict__ wv,
                                              const half_t* __restrict__ wo,
                                              float* __restrict__ y) {
  extern __shared__ char smem[];
  half_t* Xs = (half_t*)smem;                 // [NP][DD]
  half_t* Qs = Xs + NP * DD;                  // [NP][32] (cols 16..31 zero)
  half_t* Ks = Qs + NP * 32;                  // [NP][32]
  half_t* Vt = Ks + NP * 32;                  // [16][KPAD] transposed V (pad zero)
  half_t* Ps = Vt + 16 * KPAD;                // [NP][KPAD]
  half_t* HC = Ps + NP * KPAD;                // [NP][DD]

  const int tid = threadIdx.x, lane = tid & 31, wave = tid >> 5;
  const int b = blockIdx.x;
  const float* hb = h + (size_t)b * NN * DD;
  float* yb = y + (size_t)b * NN * DD;

#if HAVE_TDM
  // TDM: DMA the whole 200x128 f32 tile into the (not-yet-used) Ps/HC region,
  // then convert to f16 X. Frees VALU/VMEM pipes during the copy.
  float* stage = (float*)Ps;                  // 102.4KB staging inside 146KB region
  if (wave == 0) tdm_load_1d(hb, stage, NN * DD);
  for (int idx = tid; idx < NP * 32; idx += 256) { Qs[idx] = (half_t)0.f; Ks[idx] = (half_t)0.f; }
  for (int idx = tid; idx < 16 * KPAD; idx += 256) Vt[idx] = (half_t)0.f;
  __syncthreads();
  for (int q4 = tid; q4 < NP * DD / 4; q4 += 256) {
    int base = q4 * 4;
    int row  = base / DD;
    if (row < NN) {
      float4 v = ((const float4*)stage)[q4];
      Xs[base + 0] = (half_t)v.x; Xs[base + 1] = (half_t)v.y;
      Xs[base + 2] = (half_t)v.z; Xs[base + 3] = (half_t)v.w;
    } else {
      Xs[base + 0] = (half_t)0.f; Xs[base + 1] = (half_t)0.f;
      Xs[base + 2] = (half_t)0.f; Xs[base + 3] = (half_t)0.f;
    }
  }
#else
  for (int q4 = tid; q4 < NP * DD / 4; q4 += 256) {
    int base = q4 * 4;
    int row  = base / DD;
    if (row < NN) {
      float4 v = ((const float4*)hb)[q4];
      Xs[base + 0] = (half_t)v.x; Xs[base + 1] = (half_t)v.y;
      Xs[base + 2] = (half_t)v.z; Xs[base + 3] = (half_t)v.w;
    } else {
      Xs[base + 0] = (half_t)0.f; Xs[base + 1] = (half_t)0.f;
      Xs[base + 2] = (half_t)0.f; Xs[base + 3] = (half_t)0.f;
    }
  }
  for (int idx = tid; idx < NP * 32; idx += 256) { Qs[idx] = (half_t)0.f; Ks[idx] = (half_t)0.f; }
  for (int idx = tid; idx < 16 * KPAD; idx += 256) Vt[idx] = (half_t)0.f;
#endif

  for (int hh = 0; hh < HH; ++hh) {
    __syncthreads();
    // ---- Q projection ----
    {
      const half_t* W = wq + (size_t)hh * DD * KK;
      v16h bw[4];
#pragma unroll
      for (int kt = 0; kt < 4; ++kt) bw[kt] = frag_bp(W, 1, kt, 0, lane);
      for (int mt = wave; mt < 13; mt += 8) {
        v8f c = {};
#pragma unroll
        for (int kt = 0; kt < 4; ++kt) c = wmma16(frag_a(Xs, DD, mt * 16, kt * 32, lane), bw[kt], c);
        store_c_half(Qs, 32, mt * 16, 0, lane, c, 0.25f);   // fold 1/sqrt(KD)
      }
    }
    // ---- K projection ----
    {
      const half_t* W = wk + (size_t)hh * DD * KK;
      v16h bw[4];
#pragma unroll
      for (int kt = 0; kt < 4; ++kt) bw[kt] = frag_bp(W, 1, kt, 0, lane);
      for (int mt = wave; mt < 13; mt += 8) {
        v8f c = {};
#pragma unroll
        for (int kt = 0; kt < 4; ++kt) c = wmma16(frag_a(Xs, DD, mt * 16, kt * 32, lane), bw[kt], c);
        store_c_half(Ks, 32, mt * 16, 0, lane, c, 1.0f);
      }
    }
    // ---- V projection (stored transposed) ----
    {
      const half_t* W = wv + (size_t)hh * DD * KK;
      v16h bw[4];
#pragma unroll
      for (int kt = 0; kt < 4; ++kt) bw[kt] = frag_bp(W, 1, kt, 0, lane);
      for (int mt = wave; mt < 13; mt += 8) {
        v8f c = {};
#pragma unroll
        for (int kt = 0; kt < 4; ++kt) c = wmma16(frag_a(Xs, DD, mt * 16, kt * 32, lane), bw[kt], c);
        store_c_half_T(Vt, KPAD, mt * 16, 0, lane, c);
      }
    }
    __syncthreads();
    // ---- S = Q @ K^T ----
    for (int mt = wave; mt < 13; mt += 8) {
      v16h a = frag_a(Qs, 32, mt * 16, 0, lane);
#pragma unroll
      for (int nt = 0; nt < 13; ++nt) {
        v8f c = {};
        c = wmma16(a, frag_bt16(Ks, 32, 0, nt * 16, lane), c);
        store_c_half(Ps, KPAD, mt * 16, nt * 16, lane, c, 1.0f);
      }
    }
    __syncthreads();
    // ---- softmax ----
    for (int r = wave; r < NP; r += 8) {
      half_t* row = Ps + (size_t)r * KPAD;
      float m = -1e30f;
      for (int cidx = lane; cidx < NN; cidx += 32) m = fmaxf(m, (float)row[cidx]);
      m = wave_max(m);
      float s = 0.0f;
      for (int cidx = lane; cidx < NN; cidx += 32) {
        float e = __expf((float)row[cidx] - m);
        row[cidx] = (half_t)e;
        s += e;
      }
      s = wave_sum(s);
      float inv = 1.0f / s;
      for (int cidx = lane; cidx < KPAD; cidx += 32)
        row[cidx] = (cidx < NN) ? (half_t)((float)row[cidx] * inv) : (half_t)0.f;
    }
    __syncthreads();
    // ---- head output = P @ V ----
    {
      v16h vb[7];
#pragma unroll
      for (int kt = 0; kt < 7; ++kt) vb[kt] = frag_bt16(Vt, KPAD, kt * 32, 0, lane);
      for (int mt = wave; mt < 13; mt += 8) {
        v8f c = {};
#pragma unroll
        for (int kt = 0; kt < 7; ++kt) c = wmma16(frag_a(Ps, KPAD, mt * 16, kt * 32, lane), vb[kt], c);
        store_c_half(HC, DD, mt * 16, hh * 16, lane, c, 1.0f);
      }
    }
  }
  __syncthreads();
  // ---- output projection + residual ----
  for (int nt = 0; nt < 8; ++nt) {
    v16h bw[4];
#pragma unroll
    for (int kt = 0; kt < 4; ++kt) bw[kt] = frag_bp(wo, 8, kt, nt, lane);
    for (int mt = wave; mt < 13; mt += 8) {
      v8f c = {};
#pragma unroll
      for (int kt = 0; kt < 4; ++kt) c = wmma16(frag_a(HC, DD, mt * 16, kt * 32, lane), bw[kt], c);
      int row0 = mt * 16 + ((lane & 16) ? 8 : 0);
      int col  = nt * 16 + (lane & 15);
#pragma unroll
      for (int r = 0; r < 8; ++r) {
        int row = row0 + r;
        if (row < NN) {
          size_t i = (size_t)row * DD + col;
          yb[i] = c[r] + hb[i];
        }
      }
    }
  }
}

// ---------------- batch-norm statistics ----------------

__global__ __launch_bounds__(256) void k_stats(const float* __restrict__ y, float* __restrict__ stats,
                                               int rows_per_block) {
  int d = threadIdx.x & 127, hf = threadIdx.x >> 7;
  int r0 = blockIdx.x * rows_per_block;
  float s = 0.0f, ss = 0.0f;
  for (int r = r0 + hf; r < r0 + rows_per_block; r += 2) {
    float v = y[(size_t)r * DD + d];
    s += v; ss += v * v;
  }
  __shared__ float ls[256], lq[256];
  ls[threadIdx.x] = s; lq[threadIdx.x] = ss;
  __syncthreads();
  if (threadIdx.x < 128) {
    atomicAdd(&stats[d],       ls[d] + ls[d + 128]);
    atomicAdd(&stats[128 + d], lq[d] + lq[d + 128]);
  }
}

__global__ void k_finstats(const float* __restrict__ stats, float* __restrict__ minv) {
  int d = threadIdx.x;
  const float cnt = (float)ROWS;
  float m = stats[d] / cnt;
  float v = stats[128 + d] / cnt - m * m;
  minv[d] = m;
  minv[128 + d] = rsqrtf(v + 1e-5f);
}

// ---------------- BN1 + feedforward + residual (16 rows per block, WMMA) ----------------

__global__ __launch_bounds__(256) void k_ffn(float* __restrict__ y,
                                             const float* __restrict__ minv,
                                             const float* __restrict__ g1, const float* __restrict__ b1,
                                             const half_t* __restrict__ f1w, const float* __restrict__ f1b,
                                             const half_t* __restrict__ f2w, const float* __restrict__ f2b) {
  __shared__ half_t x1h[16 * DD];
  __shared__ float  x1f[16 * DD];
  __shared__ half_t T[16 * FFD];
#if HAVE_TDM
  __shared__ float  stage[16 * DD];
#endif
  const int tid = threadIdx.x, lane = tid & 31, wave = tid >> 5;
  const size_t r0 = (size_t)blockIdx.x * 16;

#if HAVE_TDM
  if (wave == 0) tdm_load_1d(y + r0 * DD, stage, 16 * DD);
  __syncthreads();
  for (int idx = tid; idx < 16 * DD; idx += 256) {
    int d = idx & 127;
    float v = (stage[idx] - minv[d]) * minv[128 + d] * g1[d] + b1[d];
    x1f[idx] = v;
    x1h[idx] = (half_t)v;
  }
#else
  for (int idx = tid; idx < 16 * DD; idx += 256) {
    int d = idx & 127;
    float v = (y[r0 * DD + idx] - minv[d]) * minv[128 + d] * g1[d] + b1[d];
    x1f[idx] = v;
    x1h[idx] = (half_t)v;
  }
#endif
  __syncthreads();
  // ff1
  {
    v16h a4[4];
#pragma unroll
    for (int kt = 0; kt < 4; ++kt) a4[kt] = frag_a(x1h, DD, 0, kt * 32, lane);
    for (int nt = wave; nt < 32; nt += 8) {
      v8f c = {};
#pragma unroll
      for (int kt = 0; kt < 4; ++kt) c = wmma16(a4[kt], frag_bp(f1w, 32, kt, nt, lane), c);
      int col = nt * 16 + (lane & 15);
      int row0 = (lane & 16) ? 8 : 0;
      float bias = f1b[col];
#pragma unroll
      for (int r = 0; r < 8; ++r)
        T[(size_t)(row0 + r) * FFD + col] = (half_t)fmaxf(c[r] + bias, 0.0f);
    }
  }
  __syncthreads();
  // ff2
  for (int nt = wave; nt < 8; nt += 8) {
    v8f c = {};
#pragma unroll
    for (int kt = 0; kt < 16; ++kt)
      c = wmma16(frag_a(T, FFD, 0, kt * 32, lane), frag_bp(f2w, 8, kt, nt, lane), c);
    int col = nt * 16 + (lane & 15);
    int row0 = (lane & 16) ? 8 : 0;
    float bias = f2b[col];
#pragma unroll
    for (int r = 0; r < 8; ++r) {
      int row = row0 + r;
      y[(r0 + row) * DD + col] = c[r] + bias + x1f[row * DD + col];
    }
  }
}

__global__ void k_bnapply(const float* __restrict__ y, const float* __restrict__ minv,
                          const float* __restrict__ g, const float* __restrict__ bb,
                          float* __restrict__ out) {
  size_t i = (size_t)blockIdx.x * 256 + threadIdx.x;
  int d = (int)(i & 127);
  out[i] = (y[i] - minv[d]) * minv[128 + d] * g[d] + bb[d];
}

// ---------------- precompute: fixed context + node projections ----------------

__global__ void k_fixed(const float* __restrict__ h, const float* __restrict__ w_fixed,
                        float* __restrict__ fixedc) {
  int b = blockIdx.x, d = threadIdx.x;
  __shared__ float mv[DD];
  const float* hb = h + (size_t)b * NN * DD;
  float s = 0.0f;
  for (int n = 0; n < NN; ++n) s += hb[(size_t)n * DD + d];
  mv[d] = s * (1.0f / (float)NN);
  __syncthreads();
  float o = 0.0f;
  for (int i = 0; i < DD; ++i) o += mv[i] * w_fixed[(size_t)i * DD + d];
  fixedc[(size_t)b * DD + d] = o;
}

__global__ __launch_bounds__(256) void k_proj(const float* __restrict__ h,
                                              const half_t* __restrict__ wp,
                                              float* __restrict__ proj) {
  __shared__ half_t xh[16 * DD];
#if HAVE_TDM
  __shared__ float  stage[16 * DD];
#endif
  const int tid = threadIdx.x, lane = tid & 31, wave = tid >> 5;
  const size_t r0 = (size_t)blockIdx.x * 16;
#if HAVE_TDM
  if (wave == 0) tdm_load_1d(h + r0 * DD, stage, 16 * DD);
  __syncthreads();
  for (int idx = tid; idx < 16 * DD; idx += 256) xh[idx] = (half_t)stage[idx];
#else
  for (int idx = tid; idx < 16 * DD; idx += 256) xh[idx] = (half_t)h[r0 * DD + idx];
#endif
  __syncthreads();
  v16h a4[4];
#pragma unroll
  for (int kt = 0; kt < 4; ++kt) a4[kt] = frag_a(xh, DD, 0, kt * 32, lane);
  for (int nt = wave; nt < 24; nt += 8) {
    v8f c = {};
#pragma unroll
    for (int kt = 0; kt < 4; ++kt) c = wmma16(a4[kt], frag_bp(wp, 24, kt, nt, lane), c);
    int col = nt * 16 + (lane & 15);
    int row0 = (lane & 16) ? 8 : 0;
#pragma unroll
    for (int r = 0; r < 8; ++r)
      proj[(r0 + row0 + r) * (size_t)(3 * DD) + col] = c[r];
  }
}

// ---------------- greedy decoder: persistent block per batch element, LDS-resident ----

__global__ __launch_bounds__(256) void k_decode(const float* __restrict__ h,
                                                const float* __restrict__ proj,
                                                const float* __restrict__ fixedc,
                                                const float* __restrict__ w_step,
                                                const float* __restrict__ w_out,
                                                const float* __restrict__ demand,
                                                float* __restrict__ logp_out,
                                                int* __restrict__ pi_out) {
  extern __shared__ char dsm[];
  half_t* P16  = (half_t*)dsm;              // [200][384] gk|gv|lk (f16)
  half_t* WO16 = P16 + NN * 3 * DD;         // [128][128]
  half_t* WS16 = WO16 + DD * DD;            // [129][128]

  const int b = blockIdx.x, tid = threadIdx.x;
  const int lane = tid & 31, wave = tid >> 5;
  __shared__ float ctxv[129];
  __shared__ float q[DD], gl[DD], fq[DD], logits[NN];
  __shared__ float compat[HH][NN];
  __shared__ float rv[256];
  __shared__ int   ri[256];
  __shared__ unsigned char visited[NN];
  __shared__ int cur;
  __shared__ float used;

  const float* projb = proj + (size_t)b * NN * (3 * DD);
  const float* hb    = h + (size_t)b * NN * DD;

  for (int q4 = tid; q4 < NN * 3 * DD / 4; q4 += 256) {
    float4 v = ((const float4*)projb)[q4];
    int base = q4 * 4;
    P16[base + 0] = (half_t)v.x; P16[base + 1] = (half_t)v.y;
    P16[base + 2] = (half_t)v.z; P16[base + 3] = (half_t)v.w;
  }
  for (int q4 = tid; q4 < DD * DD / 4; q4 += 256) {
    float4 v = ((const float4*)w_out)[q4];
    int base = q4 * 4;
    WO16[base + 0] = (half_t)v.x; WO16[base + 1] = (half_t)v.y;
    WO16[base + 2] = (half_t)v.z; WO16[base + 3] = (half_t)v.w;
  }
  for (int q4 = tid; q4 < (DD + 1) * DD / 4; q4 += 256) {
    float4 v = ((const float4*)w_step)[q4];
    int base = q4 * 4;
    WS16[base + 0] = (half_t)v.x; WS16[base + 1] = (half_t)v.y;
    WS16[base + 2] = (half_t)v.z; WS16[base + 3] = (half_t)v.w;
  }
  for (int i = tid; i < NN; i += 256) visited[i] = 0;
  if (tid == 0) { visited[0] = 1; cur = 0; used = 0.0f; }
  __syncthreads();

  for (int step = 0; step < NN - 1; ++step) {
    if (tid < DD) ctxv[tid] = hb[(size_t)cur * DD + tid];
    if (tid == DD) ctxv[DD] = 1.0f - used;
    __syncthreads();
    if (tid < DD) {
      float s = fixedc[(size_t)b * DD + tid];
      for (int i = 0; i < DD + 1; ++i) s += ctxv[i] * (float)WS16[i * DD + tid];
      q[tid] = s;
    }
    __syncthreads();
    for (int idx = tid; idx < HH * NN; idx += 256) {
      int hh = idx / NN, n = idx - hh * NN;
      const half_t* pr = P16 + (size_t)n * (3 * DD) + hh * KK;
      float s = 0.0f;
#pragma unroll
      for (int k = 0; k < KK; ++k) s += q[hh * KK + k] * (float)pr[k];
      compat[hh][n] = visited[n] ? -1e30f : s * 0.25f;
    }
    __syncthreads();
    {
      float m = -1e30f;
      for (int n = lane; n < NN; n += 32) m = fmaxf(m, compat[wave][n]);
      m = wave_max(m);
      float s = 0.0f;
      for (int n = lane; n < NN; n += 32) {
        float e = __expf(compat[wave][n] - m);
        compat[wave][n] = e;
        s += e;
      }
      s = wave_sum(s);
      float inv = 1.0f / s;
      for (int n = lane; n < NN; n += 32) compat[wave][n] *= inv;
    }
    __syncthreads();
    if (tid < DD) {
      int hh = tid >> 4;
      float s = 0.0f;
      for (int n = 0; n < NN; ++n) s += compat[hh][n] * (float)P16[(size_t)n * (3 * DD) + DD + tid];
      gl[tid] = s;
    }
    __syncthreads();
    if (tid < DD) {
      float s = 0.0f;
      for (int i = 0; i < DD; ++i) s += gl[i] * (float)WO16[i * DD + tid];
      fq[tid] = s;
    }
    __syncthreads();
    if (tid < NN) {
      const half_t* pr = P16 + (size_t)tid * (3 * DD) + 2 * DD;
      float s = 0.0f;
      for (int i = 0; i < DD; ++i) s += fq[i] * (float)pr[i];
      s = tanhf(s * 0.08838834764831845f) * 10.0f;
      logits[tid] = visited[tid] ? -1e30f : s;
    }
    __syncthreads();
    rv[tid] = (tid < NN) ? logits[tid] : -3.0e30f;
    ri[tid] = (tid < NN) ? tid : 0x7fffffff;
    __syncthreads();
    for (int s = 128; s > 0; s >>= 1) {
      if (tid < s) {
        if (rv[tid + s] > rv[tid] || (rv[tid + s] == rv[tid] && ri[tid + s] < ri[tid])) {
          rv[tid] = rv[tid + s]; ri[tid] = ri[tid + s];
        }
      }
      __syncthreads();
    }
    float m = rv[0];
    int sel = ri[0];
    __syncthreads();
    rv[tid] = (tid < NN) ? __expf(logits[tid] - m) : 0.0f;
    __syncthreads();
    for (int s = 128; s > 0; s >>= 1) {
      if (tid < s) rv[tid] += rv[tid + s];
      __syncthreads();
    }
    float lse = m + logf(rv[0]);
    if (tid < NN) logp_out[((size_t)b * (NN - 1) + step) * NN + tid] = logits[tid] - lse;
    __syncthreads();
    if (tid == 0) {
      pi_out[(size_t)b * (NN - 1) + step] = sel;
      visited[sel] = 1;
      used += demand[(size_t)b * NN + sel];
      cur = sel;
    }
    __syncthreads();
  }
}

// ---------------- host ----------------

extern "C" void kernel_launch(void* const* d_in, const int* in_sizes, int n_in,
                              void* d_out, int out_size, void* d_ws, size_t ws_size,
                              hipStream_t stream) {
  const float* coords  = (const float*)d_in[0];
  const float* demand  = (const float*)d_in[1];
  const float* w_depot = (const float*)d_in[2];
  const float* b_depot = (const float*)d_in[3];
  const float* w_init  = (const float*)d_in[4];
  const float* b_init  = (const float*)d_in[5];
  const float* Wq      = (const float*)d_in[6];
  const float* Wk      = (const float*)d_in[7];
  const float* Wv      = (const float*)d_in[8];
  const float* Wo      = (const float*)d_in[9];
  const float* ff1_w   = (const float*)d_in[10];
  const float* ff1_b   = (const float*)d_in[11];
  const float* ff2_w   = (const float*)d_in[12];
  const float* ff2_b   = (const float*)d_in[13];
  const float* n1_g    = (const float*)d_in[14];
  const float* n1_b    = (const float*)d_in[15];
  const float* n2_g    = (const float*)d_in[16];
  const float* n2_b    = (const float*)d_in[17];
  const float* w_proj  = (const float*)d_in[18];
  const float* w_fixed = (const float*)d_in[19];
  const float* w_step  = (const float*)d_in[20];
  const float* w_out   = (const float*)d_in[21];

  char* ws = (char*)d_ws;
  auto alloc = [&](size_t bytes) -> char* {
    char* p = ws;
    ws += (bytes + 255) & ~(size_t)255;
    return p;
  };
  float*  h      = (float*)alloc(sizeof(float) * ROWS * DD);
  float*  y      = (float*)alloc(sizeof(float) * ROWS * DD);
  float*  proj   = (float*)alloc(sizeof(float) * ROWS * 3 * DD);
  float*  fixedc = (float*)alloc(sizeof(float) * BATCH * DD);
  float*  stats  = (float*)alloc(sizeof(float) * 256);
  float*  minv   = (float*)alloc(sizeof(float) * 256);
  half_t* wq16   = (half_t*)alloc(sizeof(half_t) * 2 * HH * DD * KK);
  half_t* wk16   = (half_t*)alloc(sizeof(half_t) * 2 * HH * DD * KK);
  half_t* wv16   = (half_t*)alloc(sizeof(half_t) * 2 * HH * DD * KK);
  half_t* wo16   = (half_t*)alloc(sizeof(half_t) * 2 * DD * DD);
  half_t* f116   = (half_t*)alloc(sizeof(half_t) * 2 * DD * FFD);
  half_t* f216   = (half_t*)alloc(sizeof(half_t) * 2 * FFD * DD);
  half_t* wp16   = (half_t*)alloc(sizeof(half_t) * DD * 3 * DD);

  auto pack = [&](const float* s, half_t* d, int K, int N, int nmats) {
    int tot = K * N * nmats;
    k_pack_b<<<(tot + 255) / 256, 256, 0, stream>>>(s, d, K, N, nmats);
  };
  pack(Wq, wq16, DD, KK, 2 * HH);
  pack(Wk, wk16, DD, KK, 2 * HH);
  pack(Wv, wv16, DD, KK, 2 * HH);
  pack(Wo, wo16, DD, DD, 2);
  pack(ff1_w, f116, DD, FFD, 2);
  pack(ff2_w, f216, FFD, DD, 2);
  pack(w_proj, wp16, DD, 3 * DD, 1);

  k_init_embed<<<ROWS, DD, 0, stream>>>(coords, demand, w_depot, b_depot, w_init, b_init, h);

  const size_t smem_attn =
      (size_t)(NP * DD + NP * 32 + NP * 32 + 16 * KPAD + NP * KPAD + NP * DD) * sizeof(half_t);

  for (int l = 0; l < 2; ++l) {
    k_attn<<<BATCH, 256, smem_attn, stream>>>(h,
        wq16 + (size_t)l * HH * DD * KK, wk16 + (size_t)l * HH * DD * KK,
        wv16 + (size_t)l * HH * DD * KK, wo16 + (size_t)l * DD * DD, y);

    hipMemsetAsync(stats, 0, 256 * sizeof(float), stream);
    k_stats<<<512, 256, 0, stream>>>(y, stats, ROWS / 512);
    k_finstats<<<1, 128, 0, stream>>>(stats, minv);

    k_ffn<<<ROWS / 16, 256, 0, stream>>>(y, minv, n1_g + l * DD, n1_b + l * DD,
        f116 + (size_t)l * DD * FFD, ff1_b + l * FFD,
        f216 + (size_t)l * FFD * DD, ff2_b + l * DD);

    hipMemsetAsync(stats, 0, 256 * sizeof(float), stream);
    k_stats<<<512, 256, 0, stream>>>(y, stats, ROWS / 512);
    k_finstats<<<1, 128, 0, stream>>>(stats, minv);

    k_bnapply<<<(ROWS * DD) / 256, 256, 0, stream>>>(y, minv, n2_g + l * DD, n2_b + l * DD, h);
  }

  k_fixed<<<BATCH, DD, 0, stream>>>(h, w_fixed, fixedc);
  k_proj<<<ROWS / 16, 256, 0, stream>>>(h, wp16, proj);

  const size_t smem_dec =
      (size_t)(NN * 3 * DD + DD * DD + (DD + 1) * DD) * sizeof(half_t);
  float* logp_out = (float*)d_out;
  int*   pi_out   = (int*)(logp_out + (size_t)BATCH * (NN - 1) * NN);
  k_decode<<<BATCH, 256, smem_dec, stream>>>(h, proj, fixedc, w_step, w_out, demand,
                                             logp_out, pi_out);
}
